// opinion_attention_60206851555514
// MI455X (gfx1250) — compile-verified, matrix operand order checked by reading, not run on hardware
//
#include <hip/hip_runtime.h>
#include <math.h>

typedef __attribute__((ext_vector_type(16))) _Float16 v16h;
typedef __attribute__((ext_vector_type(8)))  float    v8f;

#define HDIM 256
#define LDIM 2048
#define BATCH 8
#define KC    64                  // attention columns per round
#define NCH   (LDIM / KC)         // 32 rounds
#define VPAD  (HDIM + 4)          // 1040B row stride: b128-aligned, low-conflict

// ---------------------------------------------------------------------------
// gfx1250 async Global->LDS staging (ASYNCcnt-tracked, no VGPR round-trip)
// ---------------------------------------------------------------------------
__device__ __forceinline__ unsigned lds_off_of(const void* p) {
    // generic LDS address = {aperture_hi, lds_offset}; low 32 bits are the
    // workgroup-relative LDS byte address the hardware consumes.
    return (unsigned)(size_t)p;
}
__device__ __forceinline__ void async_load_b128(unsigned lds, const void* gaddr) {
    asm volatile("global_load_async_to_lds_b128 %0, %1, off"
                 :: "v"(lds), "v"(gaddr) : "memory");
}
__device__ __forceinline__ void wait_async0() {
    asm volatile("s_wait_asynccnt 0" ::: "memory");
}

// ---------------------------------------------------------------------------
// Kernel A: per-token opinion multiplier (8.0 if pos tag in OPINION_LIST)
// ---------------------------------------------------------------------------
__global__ void opat_mult_kernel(const int* __restrict__ pos_ids,
                                 float* __restrict__ mult) {
    const unsigned long long mask =
        (1ull<<19)|(1ull<<20)|(1ull<<21)|(1ull<<33)|(1ull<<34)|(1ull<<35)|
        (1ull<<41)|(1ull<<42)|(1ull<<43)|(1ull<<44)|(1ull<<45)|(1ull<<46);
    int i = blockIdx.x * blockDim.x + threadIdx.x;
    if (i < BATCH * LDIM) {
        int id = pos_ids[i];
        bool is_op = (id >= 0) && (id < 64) && ((mask >> id) & 1ull);
        mult[i] = is_op ? 8.0f : 1.0f;
    }
}

// ---------------------------------------------------------------------------
// Kernel B: s_out[row] = sum_h tanh( (X @ W^T)[row,h] + bias[h] ) * wa[h]
// WMMA f16 GEMM, f32 accumulate. 8 waves; wave w owns cols [32w, 32w+32).
// ---------------------------------------------------------------------------
__global__ void opat_proj_score_kernel(const float* __restrict__ X,
                                       const float* __restrict__ W,
                                       const float* __restrict__ bias,  // may be null
                                       const float* __restrict__ wa,
                                       float* __restrict__ s_out) {
    __shared__ __align__(16) float x_sm[16][36];   // 144B row stride: b128-aligned
    __shared__ float rowsum[16];

    const int tid  = threadIdx.x;
    const int lane = tid & 31;
    const int wave = tid >> 5;
    const int row0 = blockIdx.x * 16;
    const int ncol = wave * 32;

    if (tid < 16) rowsum[tid] = 0.0f;

    v8f acc0 = {}; v8f acc1 = {};

    const int kb   = (lane < 16) ? 0 : 8;   // wave32 16-bit layout: K base
    const int kb4  = kb >> 2;               // in float4 units
    const int arow = lane & 15;
    const int n0   = ncol + (lane & 15);
    const int n1   = n0 + 16;

    for (int k0 = 0; k0 < HDIM; k0 += 32) {
        __syncthreads();
        if (tid < 128) {                    // 16x32 tile = 128 float4, coalesced
            int r = tid >> 3, c4 = (tid & 7) * 4;
            *(float4*)&x_sm[r][c4] =
                *(const float4*)(X + (size_t)(row0 + r) * HDIM + k0 + c4);
        }
        __syncthreads();

        // Gather operands as b128 loads: {kb..kb+7, kb+16..kb+23}
        float xa[16], w0a[16], w1a[16];
        {
            const float4* xr  = (const float4*)&x_sm[arow][0];
            const float4* w0r = (const float4*)(W + (size_t)n0 * HDIM + k0);
            const float4* w1r = (const float4*)(W + (size_t)n1 * HDIM + k0);
            *(float4*)&xa[0]   = xr[kb4 + 0];  *(float4*)&xa[4]   = xr[kb4 + 1];
            *(float4*)&xa[8]   = xr[kb4 + 4];  *(float4*)&xa[12]  = xr[kb4 + 5];
            *(float4*)&w0a[0]  = w0r[kb4 + 0]; *(float4*)&w0a[4]  = w0r[kb4 + 1];
            *(float4*)&w0a[8]  = w0r[kb4 + 4]; *(float4*)&w0a[12] = w0r[kb4 + 5];
            *(float4*)&w1a[0]  = w1r[kb4 + 0]; *(float4*)&w1a[4]  = w1r[kb4 + 1];
            *(float4*)&w1a[8]  = w1r[kb4 + 4]; *(float4*)&w1a[12] = w1r[kb4 + 5];
        }
        v16h a, b0, b1;
        #pragma unroll
        for (int e = 0; e < 16; ++e) {
            a[e]  = (_Float16)xa[e];
            b0[e] = (_Float16)w0a[e];
            b1[e] = (_Float16)w1a[e];
        }
        acc0 = __builtin_amdgcn_wmma_f32_16x16x32_f16(false, a, false, b0,
                                                      (short)0, acc0, false, false);
        acc1 = __builtin_amdgcn_wmma_f32_16x16x32_f16(false, a, false, b1,
                                                      (short)0, acc1, false, false);
    }
    __syncthreads();

    const int mofs = (lane < 16) ? 0 : 8;
    float b0v = bias ? bias[n0]      : 0.0f;
    float b1v = bias ? bias[n0 + 16] : 0.0f;
    float w0v = wa[n0], w1v = wa[n0 + 16];
    #pragma unroll
    for (int v = 0; v < 8; ++v) {
        int m = v + mofs;
        atomicAdd(&rowsum[m], tanhf(acc0[v] + b0v) * w0v
                            + tanhf(acc1[v] + b1v) * w1v);
    }
    __syncthreads();
    if (tid < 16) s_out[row0 + tid] = rowsum[tid];
}

// ---------------------------------------------------------------------------
// Kernel C: fused flash-attention, double-buffered async V staging.
//   scores(i,j) = (st[i] + so[j] + ba) * base(|i-j|) * mult[j]
//   out = softmax_j(scores) @ opinion
// Grid: (L/16, B). Block 256 = 8 wave32; wave w owns H cols [32w, 32w+32).
// Chunk c+1's async Global->LDS copies run under chunk c's softmax + WMMAs.
// ---------------------------------------------------------------------------
__global__ void opat_attn_kernel(const float* __restrict__ opinion, // (B,L,H)
                                 const float* __restrict__ st,      // (B,L)
                                 const float* __restrict__ so,      // (B,L)
                                 const float* __restrict__ mult,    // (B,L)
                                 const float* __restrict__ ba,      // (1,)
                                 float* __restrict__ out) {         // (B,L,H)
    __shared__ __align__(16) float v_sm[2][KC][VPAD];   // double-buffered V
    __shared__ __align__(16) float p_sm[16][KC + 4];    // 272B stride: b128-aligned
    __shared__ float pmax_sm[16][16];                   // per-thread partial maxes
    __shared__ float psum_sm[16][16];                   // per-thread partial sums
    __shared__ float st_sm[16];
    __shared__ float m_sm[16], l_sm[16], alpha_sm[16], mnew_sm[16];

    const int tid  = threadIdx.x;
    const int lane = tid & 31;
    const int wave = tid >> 5;
    const int i0   = blockIdx.x * 16;
    const int b    = blockIdx.y;
    const float bav = ba[0];

    if (tid < 16) {
        st_sm[tid] = st[(size_t)b * LDIM + i0 + tid];
        m_sm[tid]  = -1e30f;
        l_sm[tid]  = 0.0f;
    }

    v8f acc0 = {}, acc1 = {};
    const int ncol = wave * 32;
    const int kb   = (lane < 16) ? 0 : 8;
    const int kb4  = kb >> 2;
    const int arow = lane & 15;
    const int n0   = ncol + (lane & 15);
    const int mofs = (lane < 16) ? 0 : 8;

    const int sr  = tid & 15;          // score row for this thread
    const int sq  = tid >> 4;          // score column group (0..15)
    const int sc4 = sq * 4;            // 4 score columns per thread

    // async staging map: thread -> (row stg_r + 4*it, float col stg_c)
    const int stg_c = (tid & 63) * 4;  // 16B aligned
    const int stg_r = tid >> 6;        // 0..3

    const float* vrow   = opinion + (size_t)b * LDIM * HDIM;
    const float* so_b   = so   + (size_t)b * LDIM;
    const float* mult_b = mult + (size_t)b * LDIM;

    // ---- prologue: kick off chunk 0 staging into buffer 0 ----
    #pragma unroll
    for (int it = 0; it < 16; ++it) {
        int r = stg_r + it * 4;
        async_load_b128(lds_off_of(&v_sm[0][r][stg_c]),
                        vrow + (size_t)r * HDIM + stg_c);
    }

    for (int c = 0; c < NCH; ++c) {
        const int j0  = c * KC;
        const int buf = c & 1;

        __syncthreads();   // B0: prior round's p_sm readers done
        // ---- raw scores for this chunk (+ per-thread partial max) ----
        {
            float sti  = st_sm[sr];
            float4 so4 = *(const float4*)(so_b   + j0 + sc4);
            float4 mu4 = *(const float4*)(mult_b + j0 + sc4);
            float sov[4] = {so4.x, so4.y, so4.z, so4.w};
            float muv[4] = {mu4.x, mu4.y, mu4.z, mu4.w};
            float lmax = -1e30f;
            #pragma unroll
            for (int u = 0; u < 4; ++u) {
                int cc = sc4 + u;
                int di = (i0 + sr) - (j0 + cc);
                float d    = fabsf((float)di);
                float base = (di == 0) ? 0.5f : (1.0f / __log2f(2.0f + d));
                float s    = (sti + sov[u] + bav) * base * muv[u];
                p_sm[sr][cc] = s;
                lmax = fmaxf(lmax, s);
            }
            pmax_sm[sq][sr] = lmax;
        }
        __syncthreads();   // B1
        // ---- online softmax: running max + rescale factor (16-way partials) ----
        if (tid < 16) {
            float mx = m_sm[tid];
            #pragma unroll
            for (int q = 0; q < 16; ++q) mx = fmaxf(mx, pmax_sm[q][tid]);
            alpha_sm[tid] = __expf(m_sm[tid] - mx);
            mnew_sm[tid]  = mx;
            m_sm[tid]     = mx;
        }
        __syncthreads();   // B2
        // ---- exponentiate in place (+ per-thread partial sum) ----
        {
            float mn = mnew_sm[sr];
            float ls = 0.0f;
            #pragma unroll
            for (int u = 0; u < 4; ++u) {
                int cc = sc4 + u;
                float e = __expf(p_sm[sr][cc] - mn);
                p_sm[sr][cc] = e;
                ls += e;
            }
            psum_sm[sq][sr] = ls;
        }
        wait_async0();     // this wave's V copies for `buf` have landed
        __syncthreads();   // B3: all waves' V + P visible; safe to reuse other buf
        // ---- kick off next chunk's staging into the other buffer ----
        if (c + 1 < NCH) {
            const float* vb = vrow + (size_t)(j0 + KC) * HDIM;
            #pragma unroll
            for (int it = 0; it < 16; ++it) {
                int r = stg_r + it * 4;
                async_load_b128(lds_off_of(&v_sm[buf ^ 1][r][stg_c]),
                                vb + (size_t)r * HDIM + stg_c);
            }
        }
        // ---- l update (threads 0..15, overlaps frag build) ----
        if (tid < 16) {
            float s = 0.0f;
            #pragma unroll
            for (int q = 0; q < 16; ++q) s += psum_sm[q][tid];
            l_sm[tid] = l_sm[tid] * alpha_sm[tid] + s;
        }
        // ---- rescale accumulators ----
        #pragma unroll
        for (int v = 0; v < 8; ++v) {
            float a = alpha_sm[v + mofs];
            acc0[v] *= a;
            acc1[v] *= a;
        }
        // ---- A fragments via b128 LDS reads ----
        float pa[2][16];
        {
            const float4* pr = (const float4*)&p_sm[arow][0];
            #pragma unroll
            for (int ks = 0; ks < 2; ++ks) {
                *(float4*)&pa[ks][0]  = pr[ks * 8 + kb4 + 0];
                *(float4*)&pa[ks][4]  = pr[ks * 8 + kb4 + 1];
                *(float4*)&pa[ks][8]  = pr[ks * 8 + kb4 + 4];
                *(float4*)&pa[ks][12] = pr[ks * 8 + kb4 + 5];
            }
        }
        // ---- P·V: 4 WMMAs per wave per round ----
        #pragma unroll
        for (int ks = 0; ks < 2; ++ks) {
            v16h af, bf0, bf1;
            #pragma unroll
            for (int e = 0; e < 16; ++e) {
                int kk = ks * 32 + kb + (e < 8 ? e : e + 8);
                af[e]  = (_Float16)pa[ks][e];
                bf0[e] = (_Float16)v_sm[buf][kk][n0];
                bf1[e] = (_Float16)v_sm[buf][kk][n0 + 16];
            }
            acc0 = __builtin_amdgcn_wmma_f32_16x16x32_f16(false, af, false, bf0,
                                                          (short)0, acc0, false, false);
            acc1 = __builtin_amdgcn_wmma_f32_16x16x32_f16(false, af, false, bf1,
                                                          (short)0, acc1, false, false);
        }
        vrow += 0; // (keep vrow as batch base; addressing uses j0 explicitly)
    }
    __syncthreads();
    // ---- normalize and store ----
    #pragma unroll
    for (int v = 0; v < 8; ++v) {
        int m = v + mofs;
        float inv = 1.0f / l_sm[m];
        size_t orow = ((size_t)b * LDIM + i0 + m) * HDIM;
        out[orow + n0]      = acc0[v] * inv;
        out[orow + n0 + 16] = acc1[v] * inv;
    }
}

// ---------------------------------------------------------------------------
// Host launcher
// ---------------------------------------------------------------------------
extern "C" void kernel_launch(void* const* d_in, const int* in_sizes, int n_in,
                              void* d_out, int out_size, void* d_ws, size_t ws_size,
                              hipStream_t stream) {
    const float* opinion = (const float*)d_in[0];
    const float* text    = (const float*)d_in[1];
    const int*   pos_ids = (const int*)d_in[2];
    const float* Wt      = (const float*)d_in[3];
    const float* bt      = (const float*)d_in[4];
    const float* Wo      = (const float*)d_in[5];
    const float* wa      = (const float*)d_in[6];
    const float* ba      = (const float*)d_in[7];
    float* out = (float*)d_out;

    float* d_st   = (float*)d_ws;
    float* d_so   = d_st + (size_t)BATCH * LDIM;
    float* d_mult = d_so + (size_t)BATCH * LDIM;

    hipLaunchKernelGGL(opat_mult_kernel,
                       dim3((BATCH * LDIM + 255) / 256), dim3(256), 0, stream,
                       pos_ids, d_mult);

    hipLaunchKernelGGL(opat_proj_score_kernel,
                       dim3(BATCH * LDIM / 16), dim3(256), 0, stream,
                       text, Wt, bt, wa, d_st);

    hipLaunchKernelGGL(opat_proj_score_kernel,
                       dim3(BATCH * LDIM / 16), dim3(256), 0, stream,
                       opinion, Wo, (const float*)nullptr, wa + HDIM, d_so);

    hipLaunchKernelGGL(opat_attn_kernel,
                       dim3(LDIM / 16, BATCH), dim3(256), 0, stream,
                       opinion, d_st, d_so, d_mult, ba, out);
}
